// Quantizer_19018115187057
// MI455X (gfx1250) — matrix operand order repaired and checked by reading, not compiled, not run on hardware
//
#include <hip/hip_runtime.h>
#include <hip/hip_bf16.h>

// CDNA5 / gfx1250 fused VQ: GEMM via bf16x3 split (V_WMMA_F32_16X16X32_BF16)
// + fused per-row argmin + gather + straight-through output + deterministic MSE.
// v2: hi/lo planes stored separately in LDS so B fragments load as 2x b128
//     with zero unpack VALU; each wave computes a 32-row M-tile (2 WMMA tiles)
//     so every B fetch feeds 12 WMMAs.

typedef __attribute__((ext_vector_type(16))) __bf16 v16bf;
typedef __attribute__((ext_vector_type(8)))  float  v8f;

#define N_ROWS   131072
#define ZDIM     64
#define KCODES   2048
#define CHUNK    256                      // codes staged in LDS per iteration
#define NCHUNKS  (KCODES / CHUNK)         // 8
#define STRD     36                       // dwords per code row per plane (72 bf16, 16B aligned)
#define WAVES_PER_BLOCK 8
#define TILES    2                        // 16-row WMMA tiles per wave
#define ROWS_PER_WAVE   (16 * TILES)      // 32
#define ROWS_PER_BLOCK  (WAVES_PER_BLOCK * ROWS_PER_WAVE)   // 256
#define NBLOCKS  (N_ROWS / ROWS_PER_BLOCK)                  // 512
#define NPART    (NBLOCKS * WAVES_PER_BLOCK)                // 4096

union Frag { v16bf v; uint4 q[2]; unsigned short u[16]; };

__device__ __forceinline__ unsigned int f2bf_rne(float x) {
  unsigned int b = __float_as_uint(x);
  return (b + 0x7FFFu + ((b >> 16) & 1u)) >> 16;   // RNE bf16 bits
}

__global__ __launch_bounds__(256, 1)
void vq_main_kernel(const float* __restrict__ z, const float* __restrict__ e,
                    float* __restrict__ out, float* __restrict__ wspart)
{
  __shared__ __align__(16) unsigned int lds_hi[CHUNK * STRD];  // bf16 hi plane, 2/dword
  __shared__ __align__(16) unsigned int lds_lo[CHUNK * STRD];  // bf16 lo plane
  __shared__ float q_lds[CHUNK];                               // 0.5*||e_k||^2 (fp32)
  __shared__ int   idx_s[ROWS_PER_BLOCK];

  const int tid  = threadIdx.x;
  const int lane = tid & 31;
  const int wave = tid >> 5;
  const int hh   = lane >> 4;     // lane half
  const int ln   = lane & 15;
  const int r0   = (blockIdx.x * WAVES_PER_BLOCK + wave) * ROWS_PER_WAVE;

  // ---- A fragments: 2 tiles x 2 K-windows, bf16 hi/lo split ----
  // contiguous K order (matches B): elem i -> k = kb + 16h + i
  Frag ah[TILES][2], al[TILES][2];
  #pragma unroll
  for (int t = 0; t < TILES; ++t) {
    const float* zrow = z + (size_t)(r0 + 16 * t + ln) * ZDIM;
    #pragma unroll
    for (int w = 0; w < 2; ++w) {
      const int kb = 32 * w;
      #pragma unroll
      for (int i = 0; i < 16; ++i) {
        int k = kb + hh * 16 + i;
        float x = zrow[k];
        unsigned int hb = f2bf_rne(x);
        float hf = __uint_as_float(hb << 16);
        unsigned int lb = f2bf_rne(x - hf);
        ah[t][w].u[i] = (unsigned short)hb;
        al[t][w].u[i] = (unsigned short)lb;
      }
    }
  }

  float best[TILES * 8];
  int   bidx[TILES * 8];
  #pragma unroll
  for (int j = 0; j < TILES * 8; ++j) { best[j] = 3.0e38f; bidx[j] = 0x7fffffff; }

  for (int chunk = 0; chunk < NCHUNKS; ++chunk) {
    const int cbase = chunk * CHUNK;
    __syncthreads();                       // previous chunk fully consumed
    // ---- cooperative stage: e chunk -> split hi/lo bf16 planes in LDS ----
    {
      const float* esrc = e + (size_t)cbase * ZDIM;
      #pragma unroll 4
      for (int it = 0; it < (CHUNK * ZDIM / 2) / 256; ++it) {  // 32 iters, 2 elems each
        int p  = it * 256 + tid;                               // pair index (coalesced)
        int c  = p >> 5, kp = p & 31;
        float2 xy = *(const float2*)(esrc + (size_t)c * ZDIM + 2 * kp);
        unsigned int h0 = f2bf_rne(xy.x), h1 = f2bf_rne(xy.y);
        float hf0 = __uint_as_float(h0 << 16), hf1 = __uint_as_float(h1 << 16);
        unsigned int l0 = f2bf_rne(xy.x - hf0), l1 = f2bf_rne(xy.y - hf1);
        lds_hi[c * STRD + kp] = h0 | (h1 << 16);
        lds_lo[c * STRD + kp] = l0 | (l1 << 16);
      }
      // fp32 0.5*||e_k||^2, one code per thread
      const float* er = e + (size_t)(cbase + tid) * ZDIM;
      float s = 0.f;
      #pragma unroll
      for (int d = 0; d < ZDIM; ++d) s = fmaf(er[d], er[d], s);
      q_lds[tid] = 0.5f * s;
    }
    __syncthreads();

    if (chunk + 1 < NCHUNKS)               // warm L2 for next chunk
      __builtin_prefetch(e + (size_t)(cbase + CHUNK) * ZDIM + tid * 16, 0, 0);

    const uint4* hp = reinterpret_cast<const uint4*>(lds_hi);
    const uint4* lp = reinterpret_cast<const uint4*>(lds_lo);

    for (int blk = 0; blk < CHUNK / 16; ++blk) {
      // ---- B fragments: contiguous K -> two b128 LDS loads each, no unpack ----
      // dword addr = row*36 + 16w + 8h  ->  uint4 addr = row*9 + 4w + 2h
      const int bi = (blk * 16 + ln) * 9 + 2 * hh;
      Frag bh[2], bl[2];
      #pragma unroll
      for (int w = 0; w < 2; ++w) {
        bh[w].q[0] = hp[bi + 4 * w];
        bh[w].q[1] = hp[bi + 4 * w + 1];
        bl[w].q[0] = lp[bi + 4 * w];
        bl[w].q[1] = lp[bi + 4 * w + 1];
      }

      // ---- bf16x3: dot = zh*eh + zl*eh + zh*el, two independent acc chains ----
      v8f acc0 = {}, acc1 = {};
      acc0 = __builtin_amdgcn_wmma_f32_16x16x32_bf16(false, ah[0][0].v, false, bh[0].v, (short)0, acc0, false, false);
      acc1 = __builtin_amdgcn_wmma_f32_16x16x32_bf16(false, ah[1][0].v, false, bh[0].v, (short)0, acc1, false, false);
      acc0 = __builtin_amdgcn_wmma_f32_16x16x32_bf16(false, ah[0][1].v, false, bh[1].v, (short)0, acc0, false, false);
      acc1 = __builtin_amdgcn_wmma_f32_16x16x32_bf16(false, ah[1][1].v, false, bh[1].v, (short)0, acc1, false, false);
      acc0 = __builtin_amdgcn_wmma_f32_16x16x32_bf16(false, al[0][0].v, false, bh[0].v, (short)0, acc0, false, false);
      acc1 = __builtin_amdgcn_wmma_f32_16x16x32_bf16(false, al[1][0].v, false, bh[0].v, (short)0, acc1, false, false);
      acc0 = __builtin_amdgcn_wmma_f32_16x16x32_bf16(false, al[0][1].v, false, bh[1].v, (short)0, acc0, false, false);
      acc1 = __builtin_amdgcn_wmma_f32_16x16x32_bf16(false, al[1][1].v, false, bh[1].v, (short)0, acc1, false, false);
      acc0 = __builtin_amdgcn_wmma_f32_16x16x32_bf16(false, ah[0][0].v, false, bl[0].v, (short)0, acc0, false, false);
      acc1 = __builtin_amdgcn_wmma_f32_16x16x32_bf16(false, ah[1][0].v, false, bl[0].v, (short)0, acc1, false, false);
      acc0 = __builtin_amdgcn_wmma_f32_16x16x32_bf16(false, ah[0][1].v, false, bl[1].v, (short)0, acc0, false, false);
      acc1 = __builtin_amdgcn_wmma_f32_16x16x32_bf16(false, ah[1][1].v, false, bl[1].v, (short)0, acc1, false, false);

      // score = 0.5||e||^2 - z.e ; this lane's code = cbase + blk*16 + ln
      const float qv   = q_lds[blk * 16 + ln];
      const int   code = cbase + blk * 16 + ln;
      #pragma unroll
      for (int j = 0; j < 8; ++j) {
        float s0 = qv - acc0[j];
        if (s0 < best[j])     { best[j] = s0;     bidx[j] = code; }
        float s1 = qv - acc1[j];
        if (s1 < best[8 + j]) { best[8 + j] = s1; bidx[8 + j] = code; }
      }
    }
  }

  // ---- cross-lane argmin within each 16-lane half (tie -> smaller index) ----
  #pragma unroll
  for (int j = 0; j < TILES * 8; ++j) {
    float v = best[j]; int ix = bidx[j];
    #pragma unroll
    for (int mask = 1; mask <= 8; mask <<= 1) {
      float ov = __shfl_xor(v, mask, 32);
      int   oi = __shfl_xor(ix, mask, 32);
      if (ov < v || (ov == v && oi < ix)) { v = ov; ix = oi; }
    }
    best[j] = v; bidx[j] = ix;
  }

  // lanes 0 and 16 publish indices: slot t*8+j -> wave row 16t + j + 8*half
  if (ln == 0) {
    #pragma unroll
    for (int t = 0; t < TILES; ++t)
      #pragma unroll
      for (int j = 0; j < 8; ++j) {
        int rw = 16 * t + j + 8 * hh;
        idx_s[wave * ROWS_PER_WAVE + rw] = bidx[t * 8 + j];
        out[(size_t)N_ROWS * ZDIM + (size_t)(r0 + rw)] = (float)bidx[t * 8 + j];
      }
  }
  __syncthreads();

  // ---- epilogue: one row per lane, z_out = z + (z_q - z), loss partial ----
  {
    const int row   = r0 + lane;
    const int myidx = idx_s[wave * ROWS_PER_WAVE + lane];
    const float4* er4 = (const float4*)(e + (size_t)myidx * ZDIM);
    const float4* zr4 = (const float4*)(z + (size_t)row   * ZDIM);
    float4*       o4  = (float4*)(out + (size_t)row * ZDIM);
    float lsum = 0.f;
    #pragma unroll
    for (int c = 0; c < ZDIM / 4; ++c) {
      float4 zv = zr4[c];
      float4 qv = er4[c];
      float4 ov;
      ov.x = zv.x + (qv.x - zv.x); ov.y = zv.y + (qv.y - zv.y);
      ov.z = zv.z + (qv.z - zv.z); ov.w = zv.w + (qv.w - zv.w);
      o4[c] = ov;
      float dx = zv.x - qv.x, dy = zv.y - qv.y, dz = zv.z - qv.z, dw = zv.w - qv.w;
      lsum = fmaf(dx, dx, lsum); lsum = fmaf(dy, dy, lsum);
      lsum = fmaf(dz, dz, lsum); lsum = fmaf(dw, dw, lsum);
    }
    #pragma unroll
    for (int mask = 1; mask < 32; mask <<= 1)
      lsum += __shfl_xor(lsum, mask, 32);
    if (lane == 0)
      wspart[blockIdx.x * WAVES_PER_BLOCK + wave] = lsum;
  }
}

// Deterministic fixed-order reduction of per-wave loss partials.
__global__ __launch_bounds__(256)
void vq_loss_reduce_kernel(const float* __restrict__ wspart, float* __restrict__ out)
{
  __shared__ float red[256];
  const int t = threadIdx.x;
  float s = 0.f;
  for (int i = 0; i < NPART / 256; ++i)
    s += wspart[t * (NPART / 256) + i];
  red[t] = s;
  __syncthreads();
  for (int stride = 128; stride > 0; stride >>= 1) {
    if (t < stride) red[t] += red[t + stride];
    __syncthreads();
  }
  if (t == 0)
    out[(size_t)N_ROWS * ZDIM + N_ROWS] = red[0] / (float)((size_t)N_ROWS * ZDIM);
}

extern "C" void kernel_launch(void* const* d_in, const int* in_sizes, int n_in,
                              void* d_out, int out_size, void* d_ws, size_t ws_size,
                              hipStream_t stream) {
  const float* z = (const float*)d_in[0];   // [131072, 64] f32
  const float* e = (const float*)d_in[1];   // [2048, 64]  f32
  float* out  = (float*)d_out;              // z_out | indices(as f32) | loss
  float* part = (float*)d_ws;               // per-wave loss partials

  vq_main_kernel<<<NBLOCKS, 256, 0, stream>>>(z, e, out, part);
  vq_loss_reduce_kernel<<<1, 256, 0, stream>>>(part, out);
}
